// SelfAttention_18116172054587
// MI455X (gfx1250) — matrix operand order repaired and checked by reading, not compiled, not run on hardware
//
#include <hip/hip_runtime.h>
#include <hip/hip_bf16.h>

// ---------------------------------------------------------------------------
// Fused self-attention for MI455X (gfx1250, wave32, WMMA).
//   b=8, t=1024, e=256, h=8, full-width heads (head dim = 256).
// All matmuls via v_wmma_f32_16x16x32_f16 (f16 in, f32 accumulate).
// ---------------------------------------------------------------------------

typedef __attribute__((ext_vector_type(16))) _Float16 v16h;
typedef __attribute__((ext_vector_type(8)))  _Float16 v8h;
typedef __attribute__((ext_vector_type(8)))  float    v8f;
typedef __attribute__((ext_vector_type(4)))  float    v4f;

constexpr int B_  = 8;
constexpr int T_  = 1024;
constexpr int E_  = 256;
constexpr int H_  = 8;
constexpr int EH  = E_ * H_;   // 2048
constexpr int M_  = B_ * T_;   // 8192

// ---- WMMA fragment loaders (layouts per CDNA5 ISA 7.12.2) ------------------
// A matrix (16xK, 16-bit): lane L holds row m=L%16; with half=L/16,
//   elems 0..7  = K (half*8 + 0..7), elems 8..15 = K (16 + half*8 + 0..7).
__device__ __forceinline__ v16h wmma_load_a_f16(const _Float16* p, int half) {
  v16h r;
  const v8h lo = *(const v8h*)(p + half * 8);
  const v8h hi = *(const v8h*)(p + 16 + half * 8);
#pragma unroll
  for (int i = 0; i < 8; ++i) { r[i] = lo[i]; r[i + 8] = hi[i]; }
  return r;
}

__device__ __forceinline__ v16h wmma_load_a_f32(const float* p, int half) {
  v16h r;
#pragma unroll
  for (int i = 0; i < 8; ++i) r[i]     = (_Float16)p[half * 8 + i];
#pragma unroll
  for (int i = 0; i < 8; ++i) r[i + 8] = (_Float16)p[16 + half * 8 + i];
  return r;
}

// B matrix (Kx16, 16-bit): lane L holds column n=L%16; with half=L/16,
//   elems 0..7 = K (half*8 + 0..7), elems 8..15 = K (16 + half*8 + 0..7).
// `p` must point at column n's data stored K-contiguous (i.e. B^T row-major).
__device__ __forceinline__ v16h wmma_load_b_f16(const _Float16* p, int half) {
  v16h r;
  const v8h lo = *(const v8h*)(p + half * 8);
  const v8h hi = *(const v8h*)(p + 16 + half * 8);
#pragma unroll
  for (int i = 0; i < 8; ++i) { r[i] = lo[i]; r[i + 8] = hi[i]; }
  return r;
}

__device__ __forceinline__ v8f wmma16x16x32(v16h a, v16h b, v8f c) {
  return __builtin_amdgcn_wmma_f32_16x16x32_f16(false, a, false, b,
                                                (short)0, c, false, false);
}

// Cooperative staging of a 32(K) x 64(N) f32 weight tile into LDS as f16,
// transposed to [n][k] so B fragments are K-contiguous ds_load_b128 pairs.
// 256 threads: each loads 8 consecutive floats of one K-row (2x b128,
// coalesced: threads 0..7 cover 64 consecutive cols = 256B), converts, and
// scatters 8 b16 stores into LDS.  Branchless next-chunk prefetch (clamped).
__device__ __forceinline__ void stage_w_tile(const float* __restrict__ W,
                                             int ldw, int k0, int kmax,
                                             int col0, _Float16* Bt,
                                             int tid) {
  const int kk  = tid >> 3;          // 0..31
  const int nn0 = (tid & 7) * 8;     // 0,8,...,56
  const float* src = W + (size_t)(k0 + kk) * ldw + col0 + nn0;
  const v4f f0 = *(const v4f*)(src);
  const v4f f1 = *(const v4f*)(src + 4);
#pragma unroll
  for (int i = 0; i < 4; ++i) Bt[(nn0 + i) * 32 + kk]     = (_Float16)f0[i];
#pragma unroll
  for (int i = 0; i < 4; ++i) Bt[(nn0 + 4 + i) * 32 + kk] = (_Float16)f1[i];
  // Prefetch the next K-chunk (clamped in-bounds; speculative anyway).
  const int kpf = (k0 + 32 < kmax) ? (k0 + 32) : k0;
  __builtin_prefetch(W + (size_t)(kpf + kk) * ldw + col0 + nn0, 0, 3);
}

// ---------------------------------------------------------------------------
// Kernel 1: QKV projections.  x(8192x256) @ W(256x2048) + bias.
// Block tile 128x64, 256 threads (8 waves x 16 rows), K chunk 32.
// blockIdx.z selects {Q, K, V}.  Q,K stored (b,h,t,e) f16; V stored
// transposed (b,h,e,t) f16 so att@V B-fragments are contiguous.
// ---------------------------------------------------------------------------
__global__ __launch_bounds__(256)
void qkv_proj_kernel(const float* __restrict__ x,
                     const float* __restrict__ Wq, const float* __restrict__ bq,
                     const float* __restrict__ Wk, const float* __restrict__ bk,
                     const float* __restrict__ Wv, const float* __restrict__ bv,
                     _Float16* __restrict__ Q, _Float16* __restrict__ Kmat,
                     _Float16* __restrict__ Vt) {
  __shared__ _Float16 Bt[64 * 32];  // W tile, stored [n][k] (K-contiguous)

  const int wave  = threadIdx.x >> 5;
  const int lane  = threadIdx.x & 31;
  const int mlane = lane & 15;
  const int half  = lane >> 4;
  const int row0  = blockIdx.x * 128 + wave * 16;
  const int col0  = blockIdx.y * 64;
  const int mat   = blockIdx.z;

  const float* W    = (mat == 0) ? Wq : (mat == 1) ? Wk : Wv;
  const float* bias = (mat == 0) ? bq : (mat == 1) ? bk : bv;

  v8f acc[4];
#pragma unroll
  for (int n = 0; n < 4; ++n)
#pragma unroll
    for (int r = 0; r < 8; ++r) acc[n][r] = 0.0f;

  for (int k0 = 0; k0 < E_; k0 += 32) {
    __syncthreads();
    stage_w_tile(W, EH, k0, E_, col0, Bt, threadIdx.x);
    __syncthreads();

    const v16h a = wmma_load_a_f32(x + (size_t)(row0 + mlane) * E_ + k0, half);
    // Load all 4 B fragments first so the 8 ds_load_b128 overlap, then issue
    // the WMMAs (partial dscnt waits instead of full serialization).
    v16h b[4];
#pragma unroll
    for (int n = 0; n < 4; ++n)
      b[n] = wmma_load_b_f16(&Bt[(n * 16 + mlane) * 32], half);
#pragma unroll
    for (int n = 0; n < 4; ++n) acc[n] = wmma16x16x32(a, b[n], acc[n]);
  }

  // Epilogue: C/D layout => element r is (row0 + r + 8*half, col0 + 16n + mlane)
  // mat is block-uniform: branch once, not per store.
  float val[4][8];
#pragma unroll
  for (int n = 0; n < 4; ++n) {
    const float bc = bias[col0 + n * 16 + mlane];
#pragma unroll
    for (int r = 0; r < 8; ++r) val[n][r] = acc[n][r] + bc;
  }

  if (mat == 2) {
#pragma unroll
    for (int n = 0; n < 4; ++n) {
      const int c = col0 + n * 16 + mlane, hh = c >> 8, ee = c & 255;
#pragma unroll
      for (int r = 0; r < 8; ++r) {
        const int gr = row0 + r + 8 * half, bb = gr >> 10, tt = gr & 1023;
        Vt[(size_t)(((bb * H_ + hh) * E_) + ee) * T_ + tt] =
            (_Float16)val[n][r];
      }
    }
  } else {
    _Float16* dst = (mat == 0) ? Q : Kmat;
#pragma unroll
    for (int n = 0; n < 4; ++n) {
      const int c = col0 + n * 16 + mlane, hh = c >> 8, ee = c & 255;
#pragma unroll
      for (int r = 0; r < 8; ++r) {
        const int gr = row0 + r + 8 * half, bb = gr >> 10, tt = gr & 1023;
        dst[(size_t)(((bb * H_ + hh) * T_) + tt) * E_ + ee] =
            (_Float16)val[n][r];
      }
    }
  }
}

// ---------------------------------------------------------------------------
// Kernel 2: flash-style causal attention, one wave per 16-query tile.
// scores = (Q K^T)/16, mask j>=i, online softmax, out = P V.
// grid = (T/16, H, B), block = 32 (one wave; EXEC all ones for WMMA).
// ---------------------------------------------------------------------------
__global__ __launch_bounds__(32)
void attn_kernel(const _Float16* __restrict__ Q,
                 const _Float16* __restrict__ Kmat,
                 const _Float16* __restrict__ Vt,
                 _Float16* __restrict__ O) {
  __shared__ _Float16 Plds[16 * 32];  // probability tile, [row][key]

  const int lane  = threadIdx.x;
  const int mlane = lane & 15;
  const int half  = lane >> 4;
  const int i0    = blockIdx.x * 16;
  const int hh    = blockIdx.y;
  const int bb    = blockIdx.z;

  const _Float16* Qb = Q    + (size_t)((bb * H_ + hh) * T_) * E_;
  const _Float16* Kb = Kmat + (size_t)((bb * H_ + hh) * T_) * E_;
  const _Float16* Vb = Vt   + (size_t)((bb * H_ + hh) * E_) * T_;

  // Cache the whole 16x256 Q tile as 8 A-fragments.
  v16h Aq[8];
#pragma unroll
  for (int c = 0; c < 8; ++c)
    Aq[c] = wmma_load_a_f16(Qb + (size_t)(i0 + mlane) * E_ + c * 32, half);

  v8f acc[16];           // 16 x (16-col e tiles), f32 accumulators
  float rm[8], rl[8];    // running row max / row sum (rows r + 8*half)
#pragma unroll
  for (int n = 0; n < 16; ++n)
#pragma unroll
    for (int r = 0; r < 8; ++r) acc[n][r] = 0.0f;
#pragma unroll
  for (int r = 0; r < 8; ++r) { rm[r] = -3.0e38f; rl[r] = 0.0f; }

  const float scale = 0.0625f;  // 1/sqrt(256)

  for (int j0 = 0; j0 < i0 + 16; j0 += 32) {  // keys 32 at a time
    // ---- scores: two 16-key column tiles, K-dim = e (8 chunks of 32) ----
    v8f s[2];
#pragma unroll
    for (int n = 0; n < 2; ++n)
#pragma unroll
      for (int r = 0; r < 8; ++r) s[n][r] = 0.0f;
#pragma unroll
    for (int c = 0; c < 8; ++c) {
      v16h bk[2];
#pragma unroll
      for (int n = 0; n < 2; ++n)
        bk[n] = wmma_load_b_f16(
            Kb + (size_t)(j0 + n * 16 + mlane) * E_ + c * 32, half);
#pragma unroll
      for (int n = 0; n < 2; ++n) s[n] = wmma16x16x32(Aq[c], bk[n], s[n]);
    }

    // ---- scale + strict causal mask, per-row block max (16-lane halves) ----
    float bm[8];
#pragma unroll
    for (int r = 0; r < 8; ++r) {
      const int gi = i0 + r + 8 * half;
#pragma unroll
      for (int n = 0; n < 2; ++n) {
        const int gj = j0 + n * 16 + mlane;
        float v = s[n][r] * scale;
        if (gj >= gi) v = -3.0e38f;  // query i sees only j < i
        s[n][r] = v;
      }
      float bmax = fmaxf(s[0][r], s[1][r]);
#pragma unroll
      for (int off = 1; off < 16; off <<= 1)
        bmax = fmaxf(bmax, __shfl_xor(bmax, off, 32));
      bm[r] = bmax;
    }

    // ---- online softmax update ----
#pragma unroll
    for (int r = 0; r < 8; ++r) {
      const float mnew = fmaxf(rm[r], bm[r]);
      const float corr = __expf(rm[r] - mnew);
      rm[r] = mnew;
      float psum = 0.0f;
#pragma unroll
      for (int n = 0; n < 2; ++n) {
        const float p = __expf(s[n][r] - mnew);
        s[n][r] = p;
        psum += p;
      }
#pragma unroll
      for (int off = 1; off < 16; off <<= 1)
        psum += __shfl_xor(psum, off, 32);
      rl[r] = rl[r] * corr + psum;
#pragma unroll
      for (int n2 = 0; n2 < 16; ++n2) acc[n2][r] *= corr;
    }

    // ---- P: C-layout -> A-layout via LDS round trip ----
    __syncthreads();
#pragma unroll
    for (int r = 0; r < 8; ++r)
#pragma unroll
      for (int n = 0; n < 2; ++n)
        Plds[(r + 8 * half) * 32 + n * 16 + mlane] = (_Float16)s[n][r];
    __syncthreads();
    const v16h Ap = wmma_load_a_f16(&Plds[mlane * 32], half);

    // ---- out += P @ V  (V transposed: column e is K-contiguous) ----
    // Groups of 4: batch the global B loads, then the WMMAs.
#pragma unroll
    for (int g = 0; g < 4; ++g) {
      v16h bv[4];
#pragma unroll
      for (int q = 0; q < 4; ++q)
        bv[q] = wmma_load_b_f16(
            Vb + (size_t)((g * 4 + q) * 16 + mlane) * T_ + j0, half);
#pragma unroll
      for (int q = 0; q < 4; ++q)
        acc[g * 4 + q] = wmma16x16x32(Ap, bv[q], acc[g * 4 + q]);
    }
  }

  // ---- normalize + store attnOut as (b, t, h*e) f16 ----
#pragma unroll
  for (int r = 0; r < 8; ++r) {
    const int gi = i0 + r + 8 * half;
    const float inv = (gi != 0 && rl[r] > 0.0f) ? 1.0f / rl[r] : 0.0f;
#pragma unroll
    for (int n2 = 0; n2 < 16; ++n2)
      O[(size_t)(bb * T_ + gi) * EH + hh * E_ + n2 * 16 + mlane] =
          (_Float16)(acc[n2][r] * inv);
  }
}

// ---------------------------------------------------------------------------
// Kernel 3: output projection.  attnOut(8192x2048,f16) @ Wu(2048x256) + bu.
// Same tiling as kernel 1; f32 output.
// ---------------------------------------------------------------------------
__global__ __launch_bounds__(256)
void out_proj_kernel(const _Float16* __restrict__ A,
                     const float* __restrict__ Wu,
                     const float* __restrict__ bu,
                     float* __restrict__ out) {
  __shared__ _Float16 Bt[64 * 32];

  const int wave  = threadIdx.x >> 5;
  const int lane  = threadIdx.x & 31;
  const int mlane = lane & 15;
  const int half  = lane >> 4;
  const int row0  = blockIdx.x * 128 + wave * 16;
  const int col0  = blockIdx.y * 64;

  v8f acc[4];
#pragma unroll
  for (int n = 0; n < 4; ++n)
#pragma unroll
    for (int r = 0; r < 8; ++r) acc[n][r] = 0.0f;

  for (int k0 = 0; k0 < EH; k0 += 32) {
    __syncthreads();
    stage_w_tile(Wu, E_, k0, EH, col0, Bt, threadIdx.x);
    __syncthreads();

    const v16h a = wmma_load_a_f16(A + (size_t)(row0 + mlane) * EH + k0, half);
    v16h b[4];
#pragma unroll
    for (int n = 0; n < 4; ++n)
      b[n] = wmma_load_b_f16(&Bt[(n * 16 + mlane) * 32], half);
#pragma unroll
    for (int n = 0; n < 4; ++n) acc[n] = wmma16x16x32(a, b[n], acc[n]);
  }

#pragma unroll
  for (int n = 0; n < 4; ++n) {
    const int c = col0 + n * 16 + mlane;
    const float bc = bu[c];
#pragma unroll
    for (int r = 0; r < 8; ++r) {
      const int gr = row0 + r + 8 * half;
      out[(size_t)gr * E_ + c] = acc[n][r] + bc;
    }
  }
}

// ---------------------------------------------------------------------------
extern "C" void kernel_launch(void* const* d_in, const int* in_sizes, int n_in,
                              void* d_out, int out_size, void* d_ws,
                              size_t ws_size, hipStream_t stream) {
  const float* x  = (const float*)d_in[0];
  const float* Wq = (const float*)d_in[1];
  const float* bq = (const float*)d_in[2];
  const float* Wk = (const float*)d_in[3];
  const float* bk = (const float*)d_in[4];
  const float* Wv = (const float*)d_in[5];
  const float* bv = (const float*)d_in[6];
  const float* Wu = (const float*)d_in[7];
  const float* bu = (const float*)d_in[8];
  float* out = (float*)d_out;

  // Workspace layout (f16): Q | K | V^T | attnOut, 32 MiB each.
  const size_t seg = (size_t)B_ * H_ * T_ * E_ * sizeof(_Float16);  // 33554432
  char* ws = (char*)d_ws;
  _Float16* Q    = (_Float16*)(ws);
  _Float16* Kmat = (_Float16*)(ws + seg);
  _Float16* Vt   = (_Float16*)(ws + 2 * seg);
  _Float16* AO   = (_Float16*)(ws + 3 * seg);

  // 1) QKV projections: 8192x2048 GEMM x3 (z dimension).
  dim3 g1(M_ / 128, EH / 64, 3);
  qkv_proj_kernel<<<g1, 256, 0, stream>>>(x, Wq, bq, Wk, bk, Wv, bv,
                                          Q, Kmat, Vt);

  // 2) Causal attention, one wave per 16-query tile per (b,h).
  dim3 g2(T_ / 16, H_, B_);
  attn_kernel<<<g2, 32, 0, stream>>>(Q, Kmat, Vt, AO);

  // 3) Output projection: 8192x2048 @ 2048x256 + bias.
  dim3 g3(M_ / 128, E_ / 64);
  out_proj_kernel<<<g3, 256, 0, stream>>>(AO, Wu, bu, out);
}
